// EncoderBlock_7842610282538
// MI455X (gfx1250) — matrix operand order repaired and checked by reading, not compiled, not run on hardware
//
#include <hip/hip_runtime.h>
#include <hip/hip_bf16.h>
#include <math.h>

// ---------------------------------------------------------------------------
// CDNA5 (gfx1250) transformer encoder block.
// GEMMs: v_wmma_f32_16x16x32_bf16 + global_load_async_to_lds_b128 (ASYNCcnt)
// ---------------------------------------------------------------------------

typedef __bf16 bf16_t;
typedef __attribute__((ext_vector_type(16))) __bf16 v16bf;
typedef __attribute__((ext_vector_type(8)))  __bf16 v8bf;
typedef __attribute__((ext_vector_type(8)))  float  v8f;

#define kB   8
#define kS   1024
#define kD   512
#define kH   8
#define kHID 2048

#define BM 128
#define BN 128
#define BK 32
#define LP 40   // LDS row pitch in bf16 elems: 32 + 8 pad -> 80B (16B aligned, conflict-free)

enum { F_RELU = 1, F_OBF16 = 2 };

__device__ __forceinline__ bf16_t f2bf(float f) {
  union { float f; unsigned u; } in; in.f = f;
  unsigned u = in.u;
  if ((u & 0x7fffffffu) > 0x7f800000u) u = 0x7fc00000u;        // quiet NaN
  else u += 0x7fffu + ((u >> 16) & 1u);                        // round-to-nearest-even
  union { unsigned short s; bf16_t b; } out;
  out.s = (unsigned short)(u >> 16);
  return out.b;
}

// Async DMA: global (16B) -> LDS, no VGPR staging.  Tracked by ASYNCcnt.
// vdst VGPR = LDS byte address (low 32 bits of the generic LDS pointer),
// vaddr = 64-bit global address (GV mode), saddr = off.
__device__ __forceinline__ void async_ld_b128(unsigned lds_off, const void* gptr) {
  asm volatile("global_load_async_to_lds_b128 %0, %1, off"
               :: "v"(lds_off), "v"((unsigned long long)(uintptr_t)gptr)
               : "memory");
}
__device__ __forceinline__ void wait_async0() {
  asm volatile("s_wait_asynccnt 0x0" ::: "memory");
}

// ---------------------------------------------------------------------------
// Generic batched GEMM:  C[z] = alpha * A[z][M,K] * (Bt[z][N,K])^T (+bias) (ReLU?)
// Grid: (N/BN, M/BM, nz);  z1=z/zDiv, z2=z%zDiv select batch strides.
// All dims are multiples of the tile -> EXEC stays all-ones (WMMA requirement).
// 8 waves: 2(M) x 4(N); each wave owns 64x32 = 4x2 WMMA accumulators.
// ---------------------------------------------------------------------------
__global__ __launch_bounds__(256)
__attribute__((amdgpu_waves_per_eu(2)))
void gemm_bf16_wmma(
    const bf16_t* __restrict__ A, const bf16_t* __restrict__ Bm,
    void* __restrict__ Cv, const float* __restrict__ bias,
    int K, int lda, int ldb, int ldc,
    long long aS1, long long aS2, long long bS1, long long bS2,
    long long cS1, long long cS2, int zDiv,
    float alpha, int flags)
{
  __shared__ bf16_t sA[2][BM * LP];
  __shared__ bf16_t sB[2][BN * LP];

  const int z  = blockIdx.z;
  const int z1 = z / zDiv, z2 = z % zDiv;
  A  += aS1 * z1 + aS2 * z2;
  Bm += bS1 * z1 + bS2 * z2;
  const long long coff = cS1 * z1 + cS2 * z2;

  const int tid  = threadIdx.x;
  const int wave = tid >> 5;
  const int lane = tid & 31;
  const int lm   = lane & 15;   // row/col within 16
  const int lh   = lane >> 4;   // half-wave select

  const int blockM = blockIdx.y * BM;
  const int blockN = blockIdx.x * BN;
  const int waveM  = (wave >> 2) * 64;
  const int waveN  = (wave & 3) * 32;

  // 128x32 bf16 tile = 512 x 16B chunks; each thread DMAs 2 chunks per matrix
  const int c0row = tid >> 2;         // 0..63
  const int c0col = (tid & 3) << 3;   // elem offset {0,8,16,24}

  auto issueAsync = [&](int buf, int k0) {
#pragma unroll
    for (int i = 0; i < 2; ++i) {
      const int row = c0row + i * 64;
      async_ld_b128((unsigned)(uintptr_t)&sA[buf][row * LP + c0col],
                    A + (size_t)(blockM + row) * lda + k0 + c0col);
      async_ld_b128((unsigned)(uintptr_t)&sB[buf][row * LP + c0col],
                    Bm + (size_t)(blockN + row) * ldb + k0 + c0col);
    }
  };

  // A fragment (16x32, ISA 7.12.2): lane(m=lm,h=lh): V0-3 K=h*8..+7, V4-7 K=16+h*8..+7
  auto fragA = [&](int buf, int mt) {
    const bf16_t* p = &sA[buf][(waveM + mt * 16 + lm) * LP + lh * 8];
    union { v16bf v; v8bf h[2]; } u;
    u.h[0] = *reinterpret_cast<const v8bf*>(p);
    u.h[1] = *reinterpret_cast<const v8bf*>(p + 16);
    return u.v;
  };
  // B fragment (32x16): lane(n=lm,h=lh): V0-3 K=h*8..+7, V4-7 K=16+h*8..+7
  auto fragB = [&](int buf, int nt) {
    const bf16_t* p = &sB[buf][(waveN + nt * 16 + lm) * LP + lh * 8];
    union { v16bf v; v8bf h[2]; } u;
    u.h[0] = *reinterpret_cast<const v8bf*>(p);
    u.h[1] = *reinterpret_cast<const v8bf*>(p + 16);
    return u.v;
  };

  v8f acc[4][2];
#pragma unroll
  for (int mt = 0; mt < 4; ++mt)
#pragma unroll
    for (int nt = 0; nt < 2; ++nt) acc[mt][nt] = (v8f)(0.0f);

  const int nk = K >> 5;
  issueAsync(0, 0);

  for (int kt = 0; kt < nk; ++kt) {
    const int cur = kt & 1;
    wait_async0();               // DMA into `cur` complete (this wave's issues)
    __syncthreads();             // all waves' DMA done + prev reads of 1-cur done
    if (kt + 1 < nk) issueAsync(1 - cur, (kt + 1) << 5);   // overlaps WMMA below

    v16bf bfr[2];
#pragma unroll
    for (int nt = 0; nt < 2; ++nt) bfr[nt] = fragB(cur, nt);
#pragma unroll
    for (int mt = 0; mt < 4; ++mt) {
      v16bf af = fragA(cur, mt);
#pragma unroll
      for (int nt = 0; nt < 2; ++nt)
        acc[mt][nt] = __builtin_amdgcn_wmma_f32_16x16x32_bf16(
            false, af, false, bfr[nt], (short)0, acc[mt][nt], false, false);
    }
  }

  // C/D layout: VGPR r -> row r + 8*lh, col = lm
  const bool relu = (flags & F_RELU) != 0;
  const bool obf  = (flags & F_OBF16) != 0;
#pragma unroll
  for (int nt = 0; nt < 2; ++nt) {
    const int n = blockN + waveN + nt * 16 + lm;
    const float bv = bias ? bias[n] : 0.0f;
#pragma unroll
    for (int mt = 0; mt < 4; ++mt) {
#pragma unroll
      for (int r = 0; r < 8; ++r) {
        const int m = blockM + waveM + mt * 16 + lh * 8 + r;
        float v = acc[mt][nt][r] * alpha + bv;
        if (relu) v = fmaxf(v, 0.0f);
        const size_t idx = (size_t)coff + (size_t)m * ldc + n;
        if (obf) reinterpret_cast<bf16_t*>(Cv)[idx] = f2bf(v);
        else     reinterpret_cast<float*>(Cv)[idx]  = v;
      }
    }
  }
}

// ---------------------------------------------------------------------------
// f32 -> bf16 downconvert (vectorized)
// ---------------------------------------------------------------------------
__global__ __launch_bounds__(256) void f32_to_bf16_k(
    const float* __restrict__ in, bf16_t* __restrict__ out, int n4)
{
  const int i = blockIdx.x * 256 + threadIdx.x;
  if (i < n4) {
    float4 v = reinterpret_cast<const float4*>(in)[i];
    out[4 * i + 0] = f2bf(v.x);
    out[4 * i + 1] = f2bf(v.y);
    out[4 * i + 2] = f2bf(v.z);
    out[4 * i + 3] = f2bf(v.w);
  }
}

// ---------------------------------------------------------------------------
// Row softmax over S=1024 f32 scores; writes bf16 probabilities IN-PLACE at the
// row base (bf16 row occupies the first half of the f32 row).  One block/row.
// ---------------------------------------------------------------------------
__global__ __launch_bounds__(256) void softmax_1024_bf16(float* __restrict__ scores)
{
  const size_t row = blockIdx.x;
  float* rp = scores + row * (size_t)kS;
  const int t = threadIdx.x;

  float4 v = reinterpret_cast<const float4*>(rp)[t];
  __shared__ float red[256];

  float mx = fmaxf(fmaxf(v.x, v.y), fmaxf(v.z, v.w));
  red[t] = mx; __syncthreads();
  for (int s = 128; s > 0; s >>= 1) { if (t < s) red[t] = fmaxf(red[t], red[t + s]); __syncthreads(); }
  mx = red[0]; __syncthreads();

  v.x = __expf(v.x - mx); v.y = __expf(v.y - mx);
  v.z = __expf(v.z - mx); v.w = __expf(v.w - mx);
  red[t] = v.x + v.y + v.z + v.w; __syncthreads();
  for (int s = 128; s > 0; s >>= 1) { if (t < s) red[t] += red[t + s]; __syncthreads(); }
  const float inv = 1.0f / red[0];

  bf16_t* op = reinterpret_cast<bf16_t*>(rp);   // safe: all row reads done pre-barrier
  op[4 * t + 0] = f2bf(v.x * inv);
  op[4 * t + 1] = f2bf(v.y * inv);
  op[4 * t + 2] = f2bf(v.z * inv);
  op[4 * t + 3] = f2bf(v.w * inv);
}

// ---------------------------------------------------------------------------
// out = LayerNorm(a + b) over D=512; optional f32 and bf16 outputs. One block/row.
// ---------------------------------------------------------------------------
__global__ __launch_bounds__(256) void add_ln_512(
    const float* __restrict__ a, const float* __restrict__ b,
    const float* __restrict__ g, const float* __restrict__ be,
    float* __restrict__ outF, bf16_t* __restrict__ outB)
{
  const size_t row = blockIdx.x;
  const int t = threadIdx.x;
  const float* ap = a + row * (size_t)kD;
  const float* bp = b + row * (size_t)kD;

  const float x0 = ap[t] + bp[t];
  const float x1 = ap[t + 256] + bp[t + 256];

  __shared__ float red[256];
  red[t] = x0 + x1; __syncthreads();
  for (int s = 128; s > 0; s >>= 1) { if (t < s) red[t] += red[t + s]; __syncthreads(); }
  const float mu = red[0] * (1.0f / kD);
  __syncthreads();

  const float d0 = x0 - mu, d1 = x1 - mu;
  red[t] = d0 * d0 + d1 * d1; __syncthreads();
  for (int s = 128; s > 0; s >>= 1) { if (t < s) red[t] += red[t + s]; __syncthreads(); }
  const float rs = rsqrtf(red[0] * (1.0f / kD) + 1e-5f);

  const float y0 = d0 * rs * g[t] + be[t];
  const float y1 = d1 * rs * g[t + 256] + be[t + 256];
  if (outF) { outF[row * kD + t] = y0; outF[row * kD + t + 256] = y1; }
  if (outB) { outB[row * kD + t] = f2bf(y0); outB[row * kD + t + 256] = f2bf(y1); }
}

// ---------------------------------------------------------------------------
static inline void launch_gemm(hipStream_t st,
    const bf16_t* A, const bf16_t* B, void* C, const float* bias,
    int M, int N, int K, int lda, int ldb, int ldc,
    long long aS1, long long aS2, long long bS1, long long bS2,
    long long cS1, long long cS2, int zDiv, int nz, float alpha, int flags)
{
  dim3 grid(N / BN, M / BM, nz), block(256);
  gemm_bf16_wmma<<<grid, block, 0, st>>>(A, B, C, bias, K, lda, ldb, ldc,
      aS1, aS2, bS1, bS2, cS1, cS2, zDiv, alpha, flags);
}

extern "C" void kernel_launch(void* const* d_in, const int* in_sizes, int n_in,
                              void* d_out, int out_size, void* d_ws, size_t ws_size,
                              hipStream_t stream)
{
  (void)in_sizes; (void)n_in; (void)out_size; (void)ws_size;

  const float* x    = (const float*)d_in[0];
  const float* Wq   = (const float*)d_in[1];
  const float* Wk   = (const float*)d_in[2];
  const float* Wv   = (const float*)d_in[3];
  const float* Wo   = (const float*)d_in[4];
  const float* bo   = (const float*)d_in[5];
  const float* ln1g = (const float*)d_in[6];
  const float* ln1b = (const float*)d_in[7];
  const float* ln2g = (const float*)d_in[8];
  const float* ln2b = (const float*)d_in[9];
  const float* W1   = (const float*)d_in[10];
  const float* b1   = (const float*)d_in[11];
  const float* W2   = (const float*)d_in[12];
  const float* b2   = (const float*)d_in[13];
  const float* W3   = (const float*)d_in[14];
  const float* b3   = (const float*)d_in[15];
  const float* W4   = (const float*)d_in[16];
  const float* b4   = (const float*)d_in[17];

  size_t off = 0;
  auto alloc = [&](size_t bytes) -> void* {
    off = (off + 255) & ~(size_t)255;
    void* p = (char*)d_ws + off;
    off += bytes;
    return p;
  };

  const size_t nX    = (size_t)kB * kS * kD;        // 4.19M
  const size_t nWh   = (size_t)kH * kD * kD;        // 2.10M
  const size_t nBHSD = (size_t)kB * kH * kS * kD;   // 33.6M
  const size_t nScr  = (size_t)kB * kH * kS * kS;   // 67.1M
  const size_t nF    = (size_t)kB * kS * kHID;      // 16.8M

  bf16_t* xb  = (bf16_t*)alloc(nX * 2);
  bf16_t* Wqb = (bf16_t*)alloc(nWh * 2);
  bf16_t* Wkb = (bf16_t*)alloc(nWh * 2);
  bf16_t* Wvb = (bf16_t*)alloc(nWh * 2);
  bf16_t* Wob = (bf16_t*)alloc(nWh * 2);                    // D x (H*D) == H*D*D elems
  bf16_t* W1b = (bf16_t*)alloc((size_t)kHID * kD * 2);
  bf16_t* W2b = (bf16_t*)alloc((size_t)kHID * kHID * 2);
  bf16_t* W3b = (bf16_t*)alloc((size_t)kHID * kHID * 2);
  bf16_t* W4b = (bf16_t*)alloc((size_t)kD * kHID * 2);
  bf16_t* qb  = (bf16_t*)alloc(nBHSD * 2);                  // [B,H,S,D]
  bf16_t* kbf = (bf16_t*)alloc(nBHSD * 2);                  // [B,H,S,D]
  bf16_t* vTb = (bf16_t*)alloc(nBHSD * 2);                  // [B,H,D,S]  (V transposed)
  float*  sc  = (float*) alloc(nScr * 4);                   // [B,H,S,S] scores / bf16 attn
  bf16_t* zc  = (bf16_t*)alloc(nBHSD * 2);                  // [B,S,H*D] head concat
  float*  ao  = (float*) alloc(nX * 4);                     // attn output proj
  float*  hF  = (float*) alloc(nX * 4);                     // LN1 out (f32, for residual)
  bf16_t* hB  = (bf16_t*)alloc(nX * 2);                     // LN1 out (bf16, FFN input)
  bf16_t* f1  = (bf16_t*)alloc(nF * 2);
  bf16_t* f2  = (bf16_t*)alloc(nF * 2);
  bf16_t* f3  = (bf16_t*)alloc(nF * 2);
  float*  f4  = (float*) alloc(nX * 4);

  auto conv = [&](const float* s, bf16_t* d, size_t n) {
    const int n4 = (int)(n >> 2);
    f32_to_bf16_k<<<dim3((n4 + 255) / 256), dim3(256), 0, stream>>>(s, d, n4);
  };
  conv(x, xb, nX);
  conv(Wq, Wqb, nWh); conv(Wk, Wkb, nWh); conv(Wv, Wvb, nWh); conv(Wo, Wob, nWh);
  conv(W1, W1b, (size_t)kHID * kD);   conv(W2, W2b, (size_t)kHID * kHID);
  conv(W3, W3b, (size_t)kHID * kHID); conv(W4, W4b, (size_t)kD * kHID);

  const long long SD = (long long)kS * kD;
  const long long DD = (long long)kD * kD;
  const long long SS = (long long)kS * kS;

  // q[b,h] = x_b @ Wq_h^T        (batched over z = b*H + h)
  launch_gemm(stream, xb, Wqb, qb, nullptr, kS, kD, kD, kD, kD, kD,
              SD, 0, 0, DD, (long long)kH * SD, SD, kH, kB * kH, 1.0f, F_OBF16);
  // k[b,h] = x_b @ Wk_h^T
  launch_gemm(stream, xb, Wkb, kbf, nullptr, kS, kD, kD, kD, kD, kD,
              SD, 0, 0, DD, (long long)kH * SD, SD, kH, kB * kH, 1.0f, F_OBF16);
  // vT[b,h] = Wv_h @ x_b^T  -> [D,S]
  launch_gemm(stream, Wvb, xb, vTb, nullptr, kD, kS, kD, kD, kD, kS,
              0, DD, SD, 0, (long long)kH * SD, SD, kH, kB * kH, 1.0f, F_OBF16);
  // scores[b,h] = q @ k^T / sqrt(D)   (f32)
  launch_gemm(stream, qb, kbf, sc, nullptr, kS, kS, kD, kD, kD, kS,
              (long long)kH * SD, SD, (long long)kH * SD, SD,
              (long long)kH * SS, SS, kH, kB * kH, 1.0f / sqrtf((float)kD), 0);
  // softmax rows -> bf16 probabilities in-place
  softmax_1024_bf16<<<dim3(kB * kH * kS), dim3(256), 0, stream>>>(sc);
  // z[b,h] = attn @ v = attn @ vT^T   -> concat layout [B,S,H*D]
  launch_gemm(stream, (const bf16_t*)sc, vTb, zc, nullptr, kS, kD, kS,
              2 * kS, kS, kH * kD,
              2LL * kH * SS, 2LL * SS, (long long)kH * SD, SD,
              (long long)kS * kH * kD, (long long)kD, kH, kB * kH, 1.0f, F_OBF16);
  // attn_out = zc @ Wo^T + bo   (f32)
  launch_gemm(stream, zc, Wob, ao, bo, kB * kS, kD, kH * kD, kH * kD, kH * kD, kD,
              0, 0, 0, 0, 0, 0, 1, 1, 1.0f, 0);
  // h = LN1(x + attn_out)
  add_ln_512<<<dim3(kB * kS), dim3(256), 0, stream>>>(x, ao, ln1g, ln1b, hF, hB);
  // FFN
  launch_gemm(stream, hB, W1b, f1, b1, kB * kS, kHID, kD, kD, kD, kHID,
              0, 0, 0, 0, 0, 0, 1, 1, 1.0f, F_RELU | F_OBF16);
  launch_gemm(stream, f1, W2b, f2, b2, kB * kS, kHID, kHID, kHID, kHID, kHID,
              0, 0, 0, 0, 0, 0, 1, 1, 1.0f, F_RELU | F_OBF16);
  launch_gemm(stream, f2, W3b, f3, b3, kB * kS, kHID, kHID, kHID, kHID, kHID,
              0, 0, 0, 0, 0, 0, 1, 1, 1.0f, F_RELU | F_OBF16);
  launch_gemm(stream, f3, W4b, f4, b4, kB * kS, kD, kHID, kHID, kHID, kD,
              0, 0, 0, 0, 0, 0, 1, 1, 1.0f, 0);
  // out = LN2(h + f4)  (f32, matches reference output dtype)
  add_ln_512<<<dim3(kB * kS), dim3(256), 0, stream>>>(hF, f4, ln2g, ln2b,
                                                      (float*)d_out, nullptr);
}
